// FlexNonLinearReadoutBlock_1778116461344
// MI455X (gfx1250) — matrix operand order repaired and checked by reading, not compiled, not run on hardware
//
#include <hip/hip_runtime.h>

// ---------------- problem constants ----------------
#define NNODES  262144
#define SIN     128
#define NVEC    64
#define SHID    192
#define SOUTC   128
#define OUTS    320          // 128 scalars + 64*3 vectors
#define INV_S   0.0883883476483184f   // 1/sqrt(128)
#define INV_V   0.125f                // 1/sqrt(64)

typedef __attribute__((ext_vector_type(16))) _Float16 v16h;
typedef __attribute__((ext_vector_type(8)))  float    v8f;
typedef __attribute__((ext_vector_type(4)))  float    f4;

// ---------------- LDS layout (bytes) ----------------
// B-fragment-swizzled f16 weights (scale folded in):
#define OFF_WS1   0        // W1_s : 128x192 -> 24576 halfs (48KB)
#define OFF_WV1   49152    // W1_v :  64x64  ->  4096 halfs ( 8KB)
#define OFF_WS2   57344    // W2_s : 128x128 -> 16384 halfs (32KB)
#define OFF_WV2   90112    // W2_v :  64x64  ->  4096 halfs ( 8KB)
#define OFF_B1    98304    // 192 f32
#define OFF_B2    99072    // 128 f32
#define OFF_XS    99584    // per-wave A-swizzled x_s / sAct : 8 * 2048 halfs (32KB)
#define OFF_XV    132352   // per-wave A-swizzled x_v / vAct : 8 * 3072 halfs (48KB)
#define OFF_OV    181504   // per-wave out_v staging f32     : 8 * 3072 f32  (96KB)
#define SMEM_BYTES 279808

// Convert W (K x Jt, row-major f32) into f16 B-fragments:
// fragment f = jt*(K/32)+ka, within fragment lane L holds column n=L&15,
// half (v,lo) holds K = ka*32 + (v>>2)*16 + (L>>4)*8 + (v&3)*2 + lo  (matches A layout).
__device__ __forceinline__ void fillB(const float* __restrict__ W, float scale,
                                      _Float16* __restrict__ dst, int K, int Jt, int tid) {
    const int KS = K >> 5;
    const int total = K * Jt;
    for (int e = tid; e < total; e += 256) {
        int f  = e >> 9, r = e & 511;
        int L  = r >> 4, h = r & 15;
        int v  = h >> 1, lo = h & 1;
        int jt = f / KS, ka = f - jt * KS;
        int n  = L & 15, g = L >> 4;
        int Kk = ka * 32 + (v >> 2) * 16 + g * 8 + (v & 3) * 2 + lo;
        dst[e] = (_Float16)(W[Kk * Jt + jt * 16 + n] * scale);
    }
}

// Swizzled index (in halfs) of element (row m, K) in the 16-bit A-fragment
// layout: lanes 0-15 M=0..15 K in {g*8+v*2+lo | v<4} U {16+g*8+...}, g=L>>4.
__device__ __forceinline__ int aSwz(int m, int K) {
    int ka = K >> 5, kk = K & 31;
    int hi = kk >> 4, kr = kk & 15;
    int g  = kr >> 3;
    int v  = hi * 4 + ((kr & 7) >> 1);
    int lo = kk & 1;
    return ka * 512 + (g * 16 + m) * 16 + v * 2 + lo;
}

extern "C" __global__ __launch_bounds__(256, 1)
void flex_gate_mlp_kernel(const float* __restrict__ xs,  const float* __restrict__ xv,
                          const float* __restrict__ W1s, const float* __restrict__ b1,
                          const float* __restrict__ W1v, const float* __restrict__ W2s,
                          const float* __restrict__ b2,  const float* __restrict__ W2v,
                          float* __restrict__ out) {
    extern __shared__ char smem[];
    _Float16* sh = (_Float16*)smem;

    const int tid  = threadIdx.x;
    const int lane = tid & 31;
    const int wv   = tid >> 5;
    const long nb  = (long)blockIdx.x * 128 + wv * 16;   // 16 nodes per wave

    // ---- phase 0: weights -> LDS f16, B-fragment order, norm scales folded ----
    fillB(W1s, INV_S, sh + (OFF_WS1 >> 1), 128, 192, tid);
    fillB(W1v, INV_V, sh + (OFF_WV1 >> 1),  64,  64, tid);
    fillB(W2s, INV_S, sh + (OFF_WS2 >> 1), 128, 128, tid);
    fillB(W2v, INV_V, sh + (OFF_WV2 >> 1),  64,  64, tid);
    float* bl1 = (float*)(smem + OFF_B1);
    float* bl2 = (float*)(smem + OFF_B2);
    if (tid < SHID)  bl1[tid] = b1[tid];
    if (tid < SOUTC) bl2[tid] = b2[tid];

    _Float16* xsS = sh + (OFF_XS >> 1) + wv * 2048;   // later reused as sAct
    _Float16* xvS = sh + (OFF_XV >> 1) + wv * 3072;   // later reused as vAct
    float*    ovS = (float*)(smem + OFF_OV) + wv * 3072;

    // ---- phase 1: stage this wave's 16 nodes, A-fragment-swizzled f16 ----
    for (int e = lane; e < 2048; e += 32) {            // x_s : 16 x 128
        int ka = e >> 9, r = e & 511;
        int L = r >> 4, h = r & 15;
        int v = h >> 1, lo = h & 1;
        int m = L & 15, g = L >> 4;
        int kk = (v >= 4) ? (16 + g * 8 + (v - 4) * 2 + lo) : (g * 8 + v * 2 + lo);
        xsS[e] = (_Float16)xs[(nb + m) * SIN + ka * 32 + kk];
    }
    for (int e = lane; e < 3072; e += 32) {            // x_v : 3 comps of 16 x 64
        int c = e >> 10, e2 = e & 1023;
        int ka = e2 >> 9, r = e2 & 511;
        int L = r >> 4, h = r & 15;
        int v = h >> 1, lo = h & 1;
        int m = L & 15, g = L >> 4;
        int kk = (v >= 4) ? (16 + g * 8 + (v - 4) * 2 + lo) : (g * 8 + v * 2 + lo);
        int i  = ka * 32 + kk;
        xvS[e] = (_Float16)xv[(nb + m) * (NVEC * 3) + i * 3 + c];
    }
    __syncthreads();   // weights + staging visible

    const _Float16* wS1 = sh + (OFF_WS1 >> 1);
    const _Float16* wV1 = sh + (OFF_WV1 >> 1);
    const _Float16* wS2 = sh + (OFF_WS2 >> 1);
    const _Float16* wV2 = sh + (OFF_WV2 >> 1);

    const int jl = lane & 15;           // C/D: column = lane&15
    const int mb = (lane >> 4) * 8;     // C/D: row base = 8*(lane>>4), +r per VGPR

    // preload A fragments so staging LDS can be reused for activations
    v16h aS[4], aV[3][2];
#pragma unroll
    for (int ka = 0; ka < 4; ka++) aS[ka] = *(const v16h*)(xsS + ka * 512 + lane * 16);
#pragma unroll
    for (int c = 0; c < 3; c++)
#pragma unroll
        for (int ka = 0; ka < 2; ka++)
            aV[c][ka] = *(const v16h*)(xvS + c * 1024 + ka * 512 + lane * 16);

    // ---- GEMM1a (gate columns 128..191), sigmoid kept in registers ----
    v8f gacc[4];
    for (int t = 0; t < 4; t++) {
        v8f acc = {};
#pragma unroll
        for (int ka = 0; ka < 4; ka++) {
            v16h b = *(const v16h*)(wS1 + ((8 + t) * 4 + ka) * 512 + lane * 16);
            acc = __builtin_amdgcn_wmma_f32_16x16x32_f16(false, aS[ka], false, b,
                                                         (short)0, acc, false, false);
        }
        float bb = bl1[128 + t * 16 + jl];
#pragma unroll
        for (int r = 0; r < 8; r++) {
            float x = acc[r] + bb;
            gacc[t][r] = 1.0f / (1.0f + __expf(-x));
        }
    }

    // ---- GEMM1a (plain scalars 0..127) -> silu -> sAct (A-swizzled f16) ----
    _Float16* sAct = xsS;
    for (int jt = 0; jt < 8; jt++) {
        v8f acc = {};
#pragma unroll
        for (int ka = 0; ka < 4; ka++) {
            v16h b = *(const v16h*)(wS1 + (jt * 4 + ka) * 512 + lane * 16);
            acc = __builtin_amdgcn_wmma_f32_16x16x32_f16(false, aS[ka], false, b,
                                                         (short)0, acc, false, false);
        }
        int   j  = jt * 16 + jl;
        float bb = bl1[j];
#pragma unroll
        for (int r = 0; r < 8; r++) {
            float x = acc[r] + bb;
            float s = x / (1.0f + __expf(-x));       // silu
            sAct[aSwz(mb + r, j)] = (_Float16)s;
        }
    }

    // ---- GEMM1b (vectors) * gate -> vAct (A-swizzled f16) ----
    // gate accumulator C-layout matches v1 C-layout element-for-element.
    _Float16* vAct = xvS;
    for (int c = 0; c < 3; c++) {
        for (int jt = 0; jt < 4; jt++) {
            v8f acc = {};
#pragma unroll
            for (int ka = 0; ka < 2; ka++) {
                v16h b = *(const v16h*)(wV1 + (jt * 2 + ka) * 512 + lane * 16);
                acc = __builtin_amdgcn_wmma_f32_16x16x32_f16(false, aV[c][ka], false, b,
                                                             (short)0, acc, false, false);
            }
            int j = jt * 16 + jl;
#pragma unroll
            for (int r = 0; r < 8; r++)
                vAct[c * 1024 + aSwz(mb + r, j)] = (_Float16)(acc[r] * gacc[jt][r]);
        }
    }

    // ---- GEMM2: preload activated A fragments ----
    v16h a2S[4], a2V[3][2];
#pragma unroll
    for (int ka = 0; ka < 4; ka++) a2S[ka] = *(const v16h*)(sAct + ka * 512 + lane * 16);
#pragma unroll
    for (int c = 0; c < 3; c++)
#pragma unroll
        for (int ka = 0; ka < 2; ka++)
            a2V[c][ka] = *(const v16h*)(vAct + c * 1024 + ka * 512 + lane * 16);

    // out_s : direct global stores (16 contiguous floats per half-wave row)
    for (int jt = 0; jt < 8; jt++) {
        v8f acc = {};
#pragma unroll
        for (int ka = 0; ka < 4; ka++) {
            v16h b = *(const v16h*)(wS2 + (jt * 4 + ka) * 512 + lane * 16);
            acc = __builtin_amdgcn_wmma_f32_16x16x32_f16(false, a2S[ka], false, b,
                                                         (short)0, acc, false, false);
        }
        int   j  = jt * 16 + jl;
        float bb = bl2[j];
#pragma unroll
        for (int r = 0; r < 8; r++)
            out[(nb + mb + r) * OUTS + j] = acc[r] + bb;
    }

    // out_v : stage (natural layout is a stride-12B scatter) then stream the
    // staged block LDS -> HBM with the gfx1250 async engine (no VGPR roundtrip)
    for (int c = 0; c < 3; c++) {
        for (int jt = 0; jt < 4; jt++) {
            v8f acc = {};
#pragma unroll
            for (int ka = 0; ka < 2; ka++) {
                v16h b = *(const v16h*)(wV2 + (jt * 2 + ka) * 512 + lane * 16);
                acc = __builtin_amdgcn_wmma_f32_16x16x32_f16(false, a2V[c][ka], false, b,
                                                             (short)0, acc, false, false);
            }
            int j = jt * 16 + jl;
#pragma unroll
            for (int r = 0; r < 8; r++)
                ovS[(mb + r) * 192 + j * 3 + c] = acc[r];
        }
    }

    // DS writes and the async engine's LDS reads are unordered: drain DScnt first.
    asm volatile("s_wait_dscnt 0" ::: "memory");
    for (int e = lane; e < 16 * 48; e += 32) {      // 16 rows x 192 f32, b128 chunks
        int row = e / 48, q = e - row * 48;
        unsigned ldsOff = (unsigned)(OFF_OV + (wv * 3072 + row * 192 + q * 4) * 4);
        unsigned gOff   = (unsigned)(((nb + row) * OUTS + SOUTC + q * 4) * 4);
        // GVS mode: mem = SGPR_base + VGPR_offset; LDS src addr from VGPR.
        asm volatile("global_store_async_from_lds_b128 %0, %1, %2"
                     :: "v"(gOff), "v"(ldsOff), "s"(out) : "memory");
    }
    asm volatile("s_wait_asynccnt 0" ::: "memory");
}

extern "C" void kernel_launch(void* const* d_in, const int* in_sizes, int n_in,
                              void* d_out, int out_size, void* d_ws, size_t ws_size,
                              hipStream_t stream) {
    const float* xs  = (const float*)d_in[0];
    const float* xv  = (const float*)d_in[1];
    const float* W1s = (const float*)d_in[2];
    const float* b1  = (const float*)d_in[3];
    const float* W1v = (const float*)d_in[4];
    const float* W2s = (const float*)d_in[5];
    const float* b2  = (const float*)d_in[6];
    const float* W2v = (const float*)d_in[7];

    hipFuncSetAttribute((const void*)flex_gate_mlp_kernel,
                        hipFuncAttributeMaxDynamicSharedMemorySize, SMEM_BYTES);

    dim3 grid(NNODES / 128), block(256);
    flex_gate_mlp_kernel<<<grid, block, SMEM_BYTES, stream>>>(
        xs, xv, W1s, b1, W1v, W2s, b2, W2v, (float*)d_out);
}
